// Attention_80436147519543
// MI455X (gfx1250) — matrix operand order repaired and checked by reading, not compiled, not run on hardware
//
#include <hip/hip_runtime.h>
#include <hip/hip_bf16.h>

// ---------------------------------------------------------------------------
// MI455X (gfx1250) fused multi-head attention with decomposed rel-pos bias.
// f16 WMMA (v_wmma_f32_16x16x32_f16) with f32 accumulate everywhere.
// ~65 GFLOP vs ~100MB traffic @ 23.3 TB/s -> matrix-core bound. This round:
// fully-unrolled SSA double buffering in the GEMM mainloop: "load step ks+1,
// then wmma step ks" with compile-time buffer indices -> one-step prefetch
// distance (partial s_wait_loadcnt) AND no ping/pong register copies.
// ---------------------------------------------------------------------------

typedef __attribute__((ext_vector_type(16))) _Float16 v16h;
typedef __attribute__((ext_vector_type(8)))  float    v8f;

union FragA { v16h h; uint4 u[2]; };   // 16 halves = 32 bytes

#define WMMA_F16(a, b, c) \
  __builtin_amdgcn_wmma_f32_16x16x32_f16(false, (a), false, (b), (short)0, (c), false, false)

#define NH   12
#define SEQ  1024      // 32*32 tokens
#define DH   64
#define CCH  768
#define BHN  96        // B * NH

__device__ __forceinline__ void load_bfrag(FragA& f, const _Float16* p) {
  f.u[0] = *(const uint4*)(p);
  f.u[1] = *(const uint4*)(p + 16);
}
__device__ __forceinline__ void load_afrag(FragA& f, const _Float16* p) {
  f.u[0] = *(const uint4*)(p);
  f.u[1] = *(const uint4*)(p + 16);
}

// ------------------------- prep kernels ------------------------------------

__global__ void cvt_f32_to_f16_kernel(const float* __restrict__ in,
                                      _Float16* __restrict__ out, int n) {
  int i = blockIdx.x * 256 + threadIdx.x;
  if (i < n) out[i] = (_Float16)in[i];
}

// in[rows][cols] (row-major f32) -> out[cols][rows] f16
__global__ void transpose_to_f16_kernel(const float* __restrict__ in,
                                        _Float16* __restrict__ out,
                                        int rows, int cols) {
  int i = blockIdx.x * 256 + threadIdx.x;
  if (i < rows * cols) {
    int r = i / cols, c = i % cols;
    out[(size_t)c * rows + r] = (_Float16)in[i];
  }
}

// ------------------------- shared GEMM mainloop ----------------------------
// 64(M) x 16(N) per wave, K=768; 24 wmma k-steps, fully unrolled with SSA
// double buffering (buffer indices are compile-time constants after unroll).
// ap0: A + (mbase+ln)*CCH + grp*8     (A-frag: lane=row, halves klo/klo+16)
// bp : Bt + (nbase+ln)*CCH + grp*8    (B-frag: lane=col, halves klo/klo+16)

__device__ __forceinline__ void gemm_mainloop(const _Float16* __restrict__ ap0,
                                              const _Float16* __restrict__ bp,
                                              v8f (&acc)[4]) {
  FragA bf[2], af[2][4];
  load_bfrag(bf[0], bp);
#pragma unroll
  for (int t = 0; t < 4; ++t) load_afrag(af[0][t], ap0 + (size_t)t * 16 * CCH);

#pragma unroll
  for (int ks = 0; ks < 24; ++ks) {
    const int cur = ks & 1;
    const int nxt = cur ^ 1;
    if (ks < 23) {           // prefetch next k-step while current computes
      const int k = (ks + 1) * 32;
      load_bfrag(bf[nxt], bp + k);
#pragma unroll
      for (int t = 0; t < 4; ++t)
        load_afrag(af[nxt][t], ap0 + (size_t)t * 16 * CCH + k);
    }
#pragma unroll
    for (int t = 0; t < 4; ++t) acc[t] = WMMA_F16(af[cur][t].h, bf[cur].h, acc[t]);
  }
}

// ------------------------- QKV projection GEMM -----------------------------

__global__ __launch_bounds__(128)
void qkv_gemm_kernel(const _Float16* __restrict__ A,
                     const _Float16* __restrict__ Bt,
                     const float* __restrict__ bias,
                     _Float16* __restrict__ Qb,
                     _Float16* __restrict__ Kb,
                     _Float16* __restrict__ Vt) {
  const int wave = threadIdx.x >> 5;
  const int lane = threadIdx.x & 31;
  const int grp  = lane >> 4;
  const int ln   = lane & 15;
  const int mbase = blockIdx.x * 64;
  const int nbase = blockIdx.y * 64 + wave * 16;

  v8f acc[4] = {};
  gemm_mainloop(A + (size_t)(mbase + ln) * CCH + grp * 8,
                Bt + (size_t)(nbase + ln) * CCH + grp * 8, acc);

  const int f     = nbase + ln;       // output column 0..2303
  const int which = f / CCH;          // 0=q 1=k 2=v (uniform per wave tile)
  const int c     = f % CCH;
  const int head  = c >> 6;
  const int d     = c & 63;
  const float bv  = bias[f];
#pragma unroll
  for (int t = 0; t < 4; ++t) {
#pragma unroll
    for (int r = 0; r < 8; ++r) {
      int m = mbase + t * 16 + grp * 8 + r;     // global row
      int b = m >> 10, s = m & 1023;
      int bh = b * NH + head;
      _Float16 v = (_Float16)(acc[t][r] + bv);
      if (which == 0)      Qb[((size_t)bh * SEQ + s) * DH + d] = v;
      else if (which == 1) Kb[((size_t)bh * SEQ + s) * DH + d] = v;
      else                 Vt[((size_t)bh * DH + d) * SEQ + s] = v;
    }
  }
}

// ------------------------- rel-pos bias tables -----------------------------

__global__ void rel_kernel(const _Float16* __restrict__ Q,
                           const float* __restrict__ rph,
                           const float* __restrict__ rpw,
                           float* __restrict__ relh,
                           float* __restrict__ relw) {
  int i = blockIdx.x * 256 + threadIdx.x;     // [bh(7)|q(10)|is_w(1)|k(5)]
  int kidx = i & 31;
  int is_w = (i >> 5) & 1;
  int q    = (i >> 6) & 1023;
  int bh   = i >> 16;
  if (bh >= BHN) return;
  int hq = q >> 5, wq = q & 31;
  const float* tab = is_w ? (rpw + (size_t)(wq - kidx + 31) * DH)
                          : (rph + (size_t)(hq - kidx + 31) * DH);
  const _Float16* qp = Q + ((size_t)bh * SEQ + q) * DH;
  float s = 0.f;
#pragma unroll 8
  for (int d = 0; d < DH; ++d) s += (float)qp[d] * tab[d];
  float* dst = is_w ? relw : relh;
  dst[((size_t)bh * SEQ + q) * 32 + kidx] = s;
}

// ------------------------- fused flash attention ---------------------------
// Grid: (16 q-blocks, 96 head-batches), 128 threads (4 waves).
// Wave owns 16 query rows; streams 1024 keys in 32-key chunks. Per chunk:
// V frags + next chunk's K frags issued up front; QK^T wmma; softmax VALU
// overlaps in-flight loads; P repacked via per-wave LDS; PV wmma.

__global__ __launch_bounds__(128)
void attn_kernel(const _Float16* __restrict__ Q,
                 const _Float16* __restrict__ K,
                 const _Float16* __restrict__ Vt,
                 const float* __restrict__ relh,
                 const float* __restrict__ relw,
                 _Float16* __restrict__ O) {
  __shared__ float    s_relh[4][16][32];
  __shared__ float    s_relw[4][16][32];
  __shared__ _Float16 s_p[4][16][32];

  const int wave  = threadIdx.x >> 5;
  const int lane  = threadIdx.x & 31;
  const int grp   = lane >> 4;
  const int ln    = lane & 15;
  const int bh    = blockIdx.y;
  const int qbase = blockIdx.x * 64 + wave * 16;   // wave's first query row

  // stage this wave's 16x32 bias rows into its private LDS region
  {
    const float* rh = relh + ((size_t)bh * SEQ + qbase) * 32;
    const float* rw = relw + ((size_t)bh * SEQ + qbase) * 32;
    float* dh_ = (float*)s_relh[wave];
    float* dw_ = (float*)s_relw[wave];
    for (int i = lane; i < 512; i += 32) { dh_[i] = rh[i]; dw_[i] = rw[i]; }
  }
  asm volatile("s_wait_dscnt 0" ::: "memory");

  // Q fragments (dh=64 -> two K-steps), resident for the whole key loop
  FragA qf0, qf1;
  {
    const _Float16* qp = Q + ((size_t)bh * SEQ + qbase + ln) * DH;
    const int klo = grp * 16;
    qf0.u[0] = *(const uint4*)(qp + klo);
    qf0.u[1] = *(const uint4*)(qp + klo + 8);
    qf1.u[0] = *(const uint4*)(qp + 32 + klo);
    qf1.u[1] = *(const uint4*)(qp + 32 + klo + 8);
  }

  // rel_w value is fixed per (lane, key-subtile) across the whole loop
  float rwv[2][8];
#pragma unroll
  for (int t = 0; t < 2; ++t)
#pragma unroll
    for (int r = 0; r < 8; ++r)
      rwv[t][r] = s_relw[wave][grp * 8 + r][t * 16 + ln];

  v8f acc[4] = {};
  float m_run[8], lsum[8];
#pragma unroll
  for (int r = 0; r < 8; ++r) { m_run[r] = -1e30f; lsum[r] = 0.f; }

  const _Float16* kbp = K  + (size_t)bh * SEQ * DH;
  const _Float16* vbp = Vt + (size_t)bh * DH * SEQ;
  const float scale = 0.125f;    // 1/sqrt(64)

  // K fragments double buffered across chunks: kf[t-tile][d-step]
  FragA kfA[2][2], kfB[2][2];
  auto load_kfrags = [&](FragA (&dst)[2][2], int kc) {
#pragma unroll
    for (int t = 0; t < 2; ++t) {
      const _Float16* kp = kbp + (size_t)(kc * 32 + t * 16 + ln) * DH + grp * 16;
      dst[t][0].u[0] = *(const uint4*)(kp);
      dst[t][0].u[1] = *(const uint4*)(kp + 8);
      dst[t][1].u[0] = *(const uint4*)(kp + 32);
      dst[t][1].u[1] = *(const uint4*)(kp + 40);
    }
  };

  auto do_chunk = [&](int kc, FragA (&kfc)[2][2], FragA (&kfn)[2][2]) {
    const int kb = kc * 32;
    // issue V loads for this chunk + K loads for the next chunk up front
    FragA vf[4];
#pragma unroll
    for (int n = 0; n < 4; ++n) {
      const _Float16* vp = vbp + (size_t)(n * 16 + ln) * SEQ + kb + grp * 16;
      vf[n].u[0] = *(const uint4*)(vp);
      vf[n].u[1] = *(const uint4*)(vp + 8);
    }
    const int kcn = (kc + 1 < 32) ? (kc + 1) : 31;   // clamped prefetch
    load_kfrags(kfn, kcn);

    // ---- scores: 16 rows x 32 keys (two 16-key N-tiles) ----
    v8f sf[2] = {};
#pragma unroll
    for (int t = 0; t < 2; ++t) {
      sf[t] = WMMA_F16(qf0.h, kfc[t][0].h, sf[t]);
      sf[t] = WMMA_F16(qf1.h, kfc[t][1].h, sf[t]);
    }

    // ---- scale + rel-pos bias (kh == kc for the whole 32-key chunk) ----
    float rhv[8];
#pragma unroll
    for (int r = 0; r < 8; ++r) rhv[r] = s_relh[wave][grp * 8 + r][kc];
#pragma unroll
    for (int t = 0; t < 2; ++t)
#pragma unroll
      for (int r = 0; r < 8; ++r)
        sf[t][r] = sf[t][r] * scale + rhv[r] + rwv[t][r];

    // ---- online softmax (row stats per-VGPR, per 16-lane half) ----
    float alpha[8];
#pragma unroll
    for (int r = 0; r < 8; ++r) {
      float mx = fmaxf(sf[0][r], sf[1][r]);
      mx = fmaxf(mx, __shfl_xor(mx, 1, 32));
      mx = fmaxf(mx, __shfl_xor(mx, 2, 32));
      mx = fmaxf(mx, __shfl_xor(mx, 4, 32));
      mx = fmaxf(mx, __shfl_xor(mx, 8, 32));
      float mn = fmaxf(m_run[r], mx);
      alpha[r] = __expf(m_run[r] - mn);
      m_run[r] = mn;
    }
#pragma unroll
    for (int t = 0; t < 2; ++t)
#pragma unroll
      for (int r = 0; r < 8; ++r)
        sf[t][r] = __expf(sf[t][r] - m_run[r]);
#pragma unroll
    for (int r = 0; r < 8; ++r) {
      float ps = sf[0][r] + sf[1][r];
      ps += __shfl_xor(ps, 1, 32);
      ps += __shfl_xor(ps, 2, 32);
      ps += __shfl_xor(ps, 4, 32);
      ps += __shfl_xor(ps, 8, 32);
      lsum[r] = lsum[r] * alpha[r] + ps;
    }
#pragma unroll
    for (int n = 0; n < 4; ++n)
#pragma unroll
      for (int r = 0; r < 8; ++r)
        acc[n][r] *= alpha[r];

    // ---- repack P: C-fragment -> LDS -> A-fragment (16x32 f16) ----
#pragma unroll
    for (int t = 0; t < 2; ++t)
#pragma unroll
      for (int r = 0; r < 8; ++r)
        s_p[wave][grp * 8 + r][t * 16 + ln] = (_Float16)sf[t][r];
    asm volatile("s_wait_dscnt 0" ::: "memory");
    FragA pf;
    {
      const _Float16* pp = &s_p[wave][ln][0];
      const int klo = grp * 16;
      pf.u[0] = *(const uint4*)(pp + klo);
      pf.u[1] = *(const uint4*)(pp + klo + 8);
    }

    // ---- PV: acc[16 x 64] += P[16 x 32] @ V[32 x 64] ----
#pragma unroll
    for (int n = 0; n < 4; ++n) acc[n] = WMMA_F16(pf.h, vf[n].h, acc[n]);
  };

  load_kfrags(kfA, 0);
#pragma unroll 1
  for (int kc = 0; kc < 32; kc += 2) {
    do_chunk(kc, kfA, kfB);
    do_chunk(kc + 1, kfB, kfA);
  }

  // ---- normalize + store O[b*1024+s][head*64+d] f16 ----
#pragma unroll
  for (int r = 0; r < 8; ++r) lsum[r] = 1.f / lsum[r];
  const int b = bh / NH, head = bh % NH;
#pragma unroll
  for (int n = 0; n < 4; ++n) {
    const int d = head * DH + n * 16 + ln;
#pragma unroll
    for (int r = 0; r < 8; ++r) {
      int m = qbase + grp * 8 + r;
      O[((size_t)(b * SEQ + m)) * CCH + d] = (_Float16)(acc[n][r] * lsum[r]);
    }
  }
}

// ------------------------- output projection GEMM --------------------------

__global__ __launch_bounds__(128)
void out_gemm_kernel(const _Float16* __restrict__ A,    // O [8192][768]
                     const _Float16* __restrict__ Bt,   // WoutT [768][768]
                     const float* __restrict__ bias,
                     float* __restrict__ out) {
  const int wave = threadIdx.x >> 5;
  const int lane = threadIdx.x & 31;
  const int grp  = lane >> 4;
  const int ln   = lane & 15;
  const int mbase = blockIdx.x * 64;
  const int nbase = blockIdx.y * 64 + wave * 16;

  v8f acc[4] = {};
  gemm_mainloop(A + (size_t)(mbase + ln) * CCH + grp * 8,
                Bt + (size_t)(nbase + ln) * CCH + grp * 8, acc);

  const int f = nbase + ln;
  const float bv = bias[f];
#pragma unroll
  for (int t = 0; t < 4; ++t)
#pragma unroll
    for (int r = 0; r < 8; ++r)
      out[(size_t)(mbase + t * 16 + grp * 8 + r) * CCH + f] = acc[t][r] + bv;
}

// ------------------------- launch ------------------------------------------

extern "C" void kernel_launch(void* const* d_in, const int* in_sizes, int n_in,
                              void* d_out, int out_size, void* d_ws, size_t ws_size,
                              hipStream_t stream) {
  const float* x    = (const float*)d_in[0];   // (8,32,32,768)
  const float* wqkv = (const float*)d_in[1];   // (768,2304)
  const float* bqkv = (const float*)d_in[2];   // (2304,)
  const float* wout = (const float*)d_in[3];   // (768,768)
  const float* bout = (const float*)d_in[4];   // (768,)
  const float* rph  = (const float*)d_in[5];   // (63,64)
  const float* rpw  = (const float*)d_in[6];   // (63,64)
  float* out = (float*)d_out;

  const size_t M = 8192;                       // B * SEQ
  char* ws = (char*)d_ws;
  _Float16* Ax    = (_Float16*)ws; ws += M * CCH * 2;            // x -> f16
  _Float16* WqkvT = (_Float16*)ws; ws += (size_t)3 * CCH * CCH * 2;
  _Float16* WoutT = (_Float16*)ws; ws += (size_t)CCH * CCH * 2;
  _Float16* Qb    = (_Float16*)ws; ws += (size_t)BHN * SEQ * DH * 2;
  _Float16* Kb    = (_Float16*)ws; ws += (size_t)BHN * SEQ * DH * 2;
  _Float16* Vt    = (_Float16*)ws; ws += (size_t)BHN * SEQ * DH * 2;
  float*    relh  = (float*)ws;    ws += (size_t)BHN * SEQ * 32 * 4;
  float*    relw  = (float*)ws;    ws += (size_t)BHN * SEQ * 32 * 4;
  _Float16* O     = (_Float16*)ws; ws += M * CCH * 2;

  cvt_f32_to_f16_kernel<<<(int)((M * CCH + 255) / 256), 256, 0, stream>>>(
      x, Ax, (int)(M * CCH));
  transpose_to_f16_kernel<<<(CCH * 3 * CCH + 255) / 256, 256, 0, stream>>>(
      wqkv, WqkvT, CCH, 3 * CCH);
  transpose_to_f16_kernel<<<(CCH * CCH + 255) / 256, 256, 0, stream>>>(
      wout, WoutT, CCH, CCH);

  qkv_gemm_kernel<<<dim3(128, 36), 128, 0, stream>>>(Ax, WqkvT, bqkv, Qb, Kb, Vt);

  rel_kernel<<<(BHN * SEQ * 64) / 256, 256, 0, stream>>>(Qb, rph, rpw, relh, relw);

  attn_kernel<<<dim3(16, BHN), 128, 0, stream>>>(Qb, Kb, Vt, relh, relw, O);

  out_gemm_kernel<<<dim3(128, 12), 128, 0, stream>>>(O, WoutT, bout, out);
}
